// SAGEGNN_49770081026064
// MI455X (gfx1250) — compile-verified, matrix operand order checked by reading, not run on hardware
//
#include <hip/hip_runtime.h>
#include <hip/hip_bf16.h>

typedef __bf16 v16bf __attribute__((ext_vector_type(16)));
typedef float  v8f   __attribute__((ext_vector_type(8)));

#define DIN  128
#define DH   256
#define DOUT 128

static __device__ __forceinline__ void atomAdd(float* p, float v) {
    __hip_atomic_fetch_add(p, v, __ATOMIC_RELAXED, __HIP_MEMORY_SCOPE_AGENT);
}

// ---------------- zero init (deg | agg1 | agg2 are contiguous in ws) ------
__global__ void sage_zero(float* p, long long n) {
    long long i = (long long)blockIdx.x * blockDim.x + threadIdx.x;
    long long s = (long long)gridDim.x * blockDim.x;
    for (; i < n; i += s) p[i] = 0.0f;
}

// ---------------- W[K,N] f32 row-major -> WT[N,K] bf16 --------------------
__global__ void sage_wt(const float* __restrict__ W, __bf16* __restrict__ WT,
                        int K, int Ncols) {
    int i = blockIdx.x * blockDim.x + threadIdx.x;
    if (i < K * Ncols) {
        int k = i / Ncols, n = i % Ncols;
        WT[n * K + k] = (__bf16)W[i];
    }
}

// ---------------- in-degree ----------------------------------------------
__global__ void sage_degree(const int* __restrict__ ei, float* __restrict__ deg, int E) {
    int e = blockIdx.x * blockDim.x + threadIdx.x;
    if (e < E) atomAdd(&deg[ei[E + e]], 1.0f);
}

// ---------------- scatter-add: one wave per edge --------------------------
template <int D>
__global__ void sage_scatter(const int* __restrict__ ei, const float* __restrict__ feat,
                             float* __restrict__ agg, int E) {
    const int lane = threadIdx.x & 31;
    const int wid  = blockIdx.x * (blockDim.x >> 5) + (threadIdx.x >> 5);
    if (wid >= E) return;
    const int src = ei[wid];
    const int dst = ei[E + wid];
    const float* s = feat + (long long)src * D;
    float*       d = agg  + (long long)dst * D;
    constexpr int PER = D / 32;                  // 4 (D=128) or 8 (D=256)
#pragma unroll
    for (int c = 0; c < PER / 4; ++c) {
        const int off = lane * PER + c * 4;
        const float4 v = *(const float4*)(s + off);
        atomAdd(d + off + 0, v.x);
        atomAdd(d + off + 1, v.y);
        atomAdd(d + off + 2, v.z);
        atomAdd(d + off + 3, v.w);
    }
}

// ---------------- fused SAGE layer: relu?( (agg/deg)@Wl + x@Wr + b ) ------
// One wave computes a 16x16 tile via v_wmma_f32_16x16x32_bf16.
template <int K, int NC, bool RELU>
__global__ void sage_gemm(const float* __restrict__ agg, const float* __restrict__ deg,
                          const float* __restrict__ xroot,
                          const __bf16* __restrict__ WlT, const __bf16* __restrict__ WrT,
                          const float* __restrict__ bias, float* __restrict__ out,
                          int totalTiles) {
    const int lane   = threadIdx.x & 31;
    const int wid    = blockIdx.x * (blockDim.x >> 5) + (threadIdx.x >> 5);
    if (wid >= totalTiles) return;               // wave-uniform: EXEC all-1s below
    const int ntN    = NC / 16;
    const int mt     = wid / ntN;
    const int nt     = wid % ntN;
    const int laneHi = lane >> 4;
    const int l15    = lane & 15;
    const int row    = mt * 16 + l15;            // A: lane owns row M
    const int col    = nt * 16 + l15;            // B: lane owns column N

    const float scale = 1.0f / fmaxf(deg[row], 1.0f);
    const float* arow  = agg   + (long long)row * K;
    const float* xrow  = xroot + (long long)row * K;
    const __bf16* blc  = WlT   + (long long)col * K;
    const __bf16* brc  = WrT   + (long long)col * K;

    v8f c;
#pragma unroll
    for (int r = 0; r < 8; ++r) c[r] = 0.0f;

#pragma unroll
    for (int ko = 0; ko < K; ko += 32) {
        const int kb = ko + 8 * laneHi;          // ISA 16-bit A/B lane layout
        v16bf a1, a2, b1, b2;
#pragma unroll
        for (int i = 0; i < 8; ++i) {
            a1[i]     = (__bf16)(arow[kb + i]      * scale);
            a1[8 + i] = (__bf16)(arow[kb + 16 + i] * scale);
            a2[i]     = (__bf16)xrow[kb + i];
            a2[8 + i] = (__bf16)xrow[kb + 16 + i];
            b1[i]     = blc[kb + i];
            b1[8 + i] = blc[kb + 16 + i];
            b2[i]     = brc[kb + i];
            b2[8 + i] = brc[kb + 16 + i];
        }
        c = __builtin_amdgcn_wmma_f32_16x16x32_bf16(false, a1, false, b1, (short)0, c, false, false);
        c = __builtin_amdgcn_wmma_f32_16x16x32_bf16(false, a2, false, b2, (short)0, c, false, false);
    }

    const float bv = bias[col];
#pragma unroll
    for (int r = 0; r < 8; ++r) {                // C/D layout: M = r + 8*laneHi
        const int rr = mt * 16 + r + 8 * laneHi;
        float v = c[r] + bv;
        if (RELU) v = fmaxf(v, 0.0f);
        out[(long long)rr * NC + col] = v;
    }
}

extern "C" void kernel_launch(void* const* d_in, const int* in_sizes, int n_in,
                              void* d_out, int out_size, void* d_ws, size_t ws_size,
                              hipStream_t stream) {
    const float* x    = (const float*)d_in[0];
    const int*   ei   = (const int*)  d_in[1];
    const float* W1l  = (const float*)d_in[2];
    const float* b1   = (const float*)d_in[3];
    const float* W1r  = (const float*)d_in[4];
    const float* W2l  = (const float*)d_in[5];
    const float* b2   = (const float*)d_in[6];
    const float* W2r  = (const float*)d_in[7];
    float* out = (float*)d_out;

    const int N = in_sizes[0] / DIN;
    const int E = in_sizes[1] / 2;

    // workspace layout (bytes); deg|agg1|agg2 contiguous so one zero pass covers them
    char* ws = (char*)d_ws;
    size_t off = 0;
    float* deg  = (float*)(ws + off); off += (size_t)N * 4;
    float* agg1 = (float*)(ws + off); off += (size_t)N * DIN * 4;
    float* agg2 = (float*)(ws + off); off += (size_t)N * DH  * 4;
    float* h    = (float*)(ws + off); off += (size_t)N * DH  * 4;
    __bf16* W1lT = (__bf16*)(ws + off); off += (size_t)DIN * DH * 2;
    __bf16* W1rT = (__bf16*)(ws + off); off += (size_t)DIN * DH * 2;
    __bf16* W2lT = (__bf16*)(ws + off); off += (size_t)DH * DOUT * 2;
    __bf16* W2rT = (__bf16*)(ws + off); off += (size_t)DH * DOUT * 2;

    const int T = 256;                        // 8 waves / block (wave32)

    // 1) zero deg + agg1 + agg2
    const long long zn = (long long)N * (1 + DIN + DH);
    sage_zero<<<4096, T, 0, stream>>>(deg, zn);

    // 2) weights -> bf16, transposed (column-contiguous for B-fragments)
    sage_wt<<<(DIN * DH + T - 1) / T, T, 0, stream>>>(W1l, W1lT, DIN, DH);
    sage_wt<<<(DIN * DH + T - 1) / T, T, 0, stream>>>(W1r, W1rT, DIN, DH);
    sage_wt<<<(DH * DOUT + T - 1) / T, T, 0, stream>>>(W2l, W2lT, DH, DOUT);
    sage_wt<<<(DH * DOUT + T - 1) / T, T, 0, stream>>>(W2r, W2rT, DH, DOUT);

    // 3) degrees
    sage_degree<<<(E + T - 1) / T, T, 0, stream>>>(ei, deg, E);

    // 4) layer-1 aggregation: agg1 += x[src]
    sage_scatter<DIN><<<(E + 7) / 8, T, 0, stream>>>(ei, x, agg1, E);

    // 5) h = relu((agg1/deg)@W1_l + x@W1_r + b1)
    {
        const int tiles = (N / 16) * (DH / 16);
        sage_gemm<DIN, DH, true><<<(tiles + 7) / 8, T, 0, stream>>>(
            agg1, deg, x, W1lT, W1rT, b1, h, tiles);
    }

    // 6) layer-2 aggregation: agg2 += h[src]
    sage_scatter<DH><<<(E + 7) / 8, T, 0, stream>>>(ei, h, agg2, E);

    // 7) out = (agg2/deg)@W2_l + h@W2_r + b2
    {
        const int tiles = (N / 16) * (DOUT / 16);
        sage_gemm<DH, DOUT, false><<<(tiles + 7) / 8, T, 0, stream>>>(
            agg2, deg, h, W2lT, W2rT, b2, out, tiles);
    }
}
